// WormholeAttentionBlock_40948218200750
// MI455X (gfx1250) — matrix operand order repaired and checked by reading, not compiled, not run on hardware
//
#include <hip/hip_runtime.h>
#include <hip/hip_bf16.h>
#include <math.h>

typedef __attribute__((ext_vector_type(16))) _Float16 v16h;
typedef __attribute__((ext_vector_type(8)))  _Float16 v8h;
typedef __attribute__((ext_vector_type(2)))  _Float16 h2v;
typedef __attribute__((ext_vector_type(2)))  float    f32x2;
typedef __attribute__((ext_vector_type(8)))  float    v8f;

// ---- problem constants ----
constexpr int   Dm   = 768;
constexpr int   Hh   = 12;
constexpr int   Pp   = 576;
constexpr int   KWw  = 32;
constexpr int   Bb   = 8;
constexpr int   Ss   = 577;       // P + 1
constexpr int   D3   = 2304;      // 3*D
constexpr int   MLPm = 3072;
constexpr float TEMP_INV = 10.0f; // 1/0.1
constexpr float SCALEa   = 0.125f; // 64^-0.5
constexpr float NEGC     = -1000000000.0f;
constexpr int   BSr  = Bb * Ss;   // 4616

// ======================= LayerNorm =======================
__global__ __launch_bounds__(256)
void ln_kernel(const float* __restrict__ x, const float* __restrict__ g,
               const float* __restrict__ bb, float* __restrict__ y)
{
    __shared__ float red[256];
    const int row = blockIdx.x, t = threadIdx.x;
    const float* xr = x + (size_t)row * Dm;
    float s = 0.f;
    for (int c = t; c < Dm; c += 256) s += xr[c];
    red[t] = s; __syncthreads();
    for (int o = 128; o > 0; o >>= 1) { if (t < o) red[t] += red[t + o]; __syncthreads(); }
    const float mean = red[0] * (1.0f / Dm);
    __syncthreads();
    float ss = 0.f;
    for (int c = t; c < Dm; c += 256) { float d = xr[c] - mean; ss += d * d; }
    red[t] = ss; __syncthreads();
    for (int o = 128; o > 0; o >>= 1) { if (t < o) red[t] += red[t + o]; __syncthreads(); }
    const float rstd = rsqrtf(red[0] * (1.0f / Dm) + 1e-5f);
    float* yr = y + (size_t)row * Dm;
    for (int c = t; c < Dm; c += 256) yr[c] = (xr[c] - mean) * rstd * g[c] + bb[c];
}

// ======================= row L2 normalize =======================
__global__ __launch_bounds__(256)
void l2norm_kernel(float* __restrict__ y)
{
    __shared__ float red[256];
    const int row = blockIdx.x, t = threadIdx.x;
    float* yr = y + (size_t)row * Dm;
    float ss = 0.f;
    for (int c = t; c < Dm; c += 256) { float v = yr[c]; ss += v * v; }
    red[t] = ss; __syncthreads();
    for (int o = 128; o > 0; o >>= 1) { if (t < o) red[t] += red[t + o]; __syncthreads(); }
    const float inv = 1.0f / fmaxf(sqrtf(red[0]), 1e-12f);
    __syncthreads();
    for (int c = t; c < Dm; c += 256) yr[c] *= inv;
}

// ======================= tiled WMMA GEMM =======================
// C[b] = A[b](MxK,f32) * B[b](KxN, or NxK when TRANSB) -> epilogue -> f32
// Block tile 128x64, K-chunk 32, 8 waves, each wave a 32x32 patch (2x2 WMMA).
// Double-buffered LDS; layouts co-designed with the 16-bit WMMA fragment
// layouts so every fragment read is a contiguous ds_load_b128.
enum { EPI_BIAS = 0, EPI_GELU = 1, EPI_RES = 2, EPI_SCORES = 3 };

template<int EPI, bool TRANSB>
__global__ __launch_bounds__(256)
void gemm_wmma_kernel(const float* __restrict__ A, const float* __restrict__ Bm,
                      const float* __restrict__ bias, const float* __restrict__ res,
                      float* __restrict__ C, int M, int N, int K,
                      long long sA, long long sB, long long sC, long long sR)
{
    // 80-byte row stride: 20-bank stride (conflict-free across 16 rows), 16B-aligned sub-rows
    __shared__ __align__(64) _Float16 As[2][128][40];  // A tiles, row-major [m][k]
    __shared__ __align__(64) _Float16 Bs[2][64][40];   // B tiles, n-major  [n][k]

    const int tid   = threadIdx.x;
    const int lane  = tid & 31;
    const int wid   = tid >> 5;
    const int waveM = wid & 3;         // 4 waves along M (4*32 = 128)
    const int waveN = wid >> 2;        // 2 waves along N (2*32 = 64)
    const int lhalf = lane >> 4;
    const int lmod  = lane & 15;
    const int tileN = blockIdx.x, tileM = blockIdx.y, bz = blockIdx.z;

    const float* Ab = A  + (size_t)bz * sA;
    const float* Bb = Bm + (size_t)bz * sB;
    float*       Cb = C  + (size_t)bz * sC;
    const float* Rb = (EPI == EPI_RES || EPI == EPI_SCORES) ? res + (size_t)bz * sR : nullptr;

    const int mrow0 = waveM * 32 + lmod;
    const int mrow1 = mrow0 + 16;
    const int ncol0 = waveN * 32 + lmod;
    const int ncol1 = ncol0 + 16;

    // ---- stage one K-chunk into LDS buffer `buf` ----
    auto stage = [&](int buf, int kt) {
        // prefetch the chunk after next (gfx1250 global_prefetch_b8)
        if (kt + 32 < K) {
            int pr = tileM * 128 + (tid >> 1);
            if (pr < M)
                __builtin_prefetch(&Ab[(size_t)pr * K + kt + 32 + (tid & 1) * 16], 0, 1);
        }
        // A tile: b64 global loads (coalesced), packed b32 LDS stores
        #pragma unroll
        for (int i = 0; i < 8; ++i) {
            int idx = i * 256 + tid;          // 2048 f32-pairs
            int r   = idx >> 4;               // 0..127
            int kp  = (idx & 15) * 2;         // 0,2,..,30
            int grow = tileM * 128 + r;
            size_t gr = (size_t)(grow < M ? grow : M - 1);   // branchless clamp
            f32x2 v = *(const f32x2*)(Ab + gr * K + kt + kp);
            h2v p;
            p.x = (_Float16)(grow < M ? v.x : 0.0f);
            p.y = (_Float16)(grow < M ? v.y : 0.0f);
            *(h2v*)&As[buf][r][kp] = p;
        }
        // B tile into n-major layout
        if (!TRANSB) {
            // B is KxN: two coalesced-along-n reads (rows k, k+1), one packed b32 store
            #pragma unroll
            for (int i = 0; i < 4; ++i) {
                int idx = i * 256 + tid;      // 1024 k-pairs
                int n  = idx & 63;
                int k  = (idx >> 6) * 2;      // 0,2,..,30
                size_t gcol = (size_t)tileN * 64 + n;
                float v0 = Bb[(size_t)(kt + k)     * N + gcol];
                float v1 = Bb[(size_t)(kt + k + 1) * N + gcol];
                h2v p;
                p.x = (_Float16)v0;
                p.y = (_Float16)v1;
                *(h2v*)&Bs[buf][n][k] = p;
            }
        } else {
            // B is NxK: contiguous along k -> b64 reads + packed b32 stores
            #pragma unroll
            for (int i = 0; i < 4; ++i) {
                int idx = i * 256 + tid;      // 1024 f32-pairs
                int n  = idx >> 4;            // 0..63
                int kp = (idx & 15) * 2;      // 0..30
                f32x2 v = *(const f32x2*)(Bb + (size_t)(tileN * 64 + n) * K + kt + kp);
                h2v p;
                p.x = (_Float16)v.x;
                p.y = (_Float16)v.y;
                *(h2v*)&Bs[buf][n][kp] = p;
            }
        }
    };

    v8f cacc[2][2] = {};

    stage(0, 0);
    __syncthreads();

    for (int kt = 0; kt < K; kt += 32) {
        const int cur = (kt >> 5) & 1;
        if (kt + 32 < K) stage(cur ^ 1, kt + 32);   // fill other buffer while computing

        // ---- build WMMA fragments: contiguous ds_load_b128 per ISA 7.12.2 ----
        // A lane(l): row = l&15 (+16 per sub-tile), k in [8h,8h+8) U [16+8h,16+8h+8)
        v8h a0lo = *(const v8h*)&As[cur][mrow0][8 * lhalf];
        v8h a0hi = *(const v8h*)&As[cur][mrow0][16 + 8 * lhalf];
        v8h a1lo = *(const v8h*)&As[cur][mrow1][8 * lhalf];
        v8h a1hi = *(const v8h*)&As[cur][mrow1][16 + 8 * lhalf];
        // B lane(l): col = l&15 (+16), k in [8h,8h+8) U [16+8h,16+8h+8)
        v8h b0lo = *(const v8h*)&Bs[cur][ncol0][8 * lhalf];
        v8h b0hi = *(const v8h*)&Bs[cur][ncol0][16 + 8 * lhalf];
        v8h b1lo = *(const v8h*)&Bs[cur][ncol1][8 * lhalf];
        v8h b1hi = *(const v8h*)&Bs[cur][ncol1][16 + 8 * lhalf];

        v16h af[2], bf[2];
        af[0] = __builtin_shufflevector(a0lo, a0hi, 0,1,2,3,4,5,6,7,8,9,10,11,12,13,14,15);
        af[1] = __builtin_shufflevector(a1lo, a1hi, 0,1,2,3,4,5,6,7,8,9,10,11,12,13,14,15);
        bf[0] = __builtin_shufflevector(b0lo, b0hi, 0,1,2,3,4,5,6,7,8,9,10,11,12,13,14,15);
        bf[1] = __builtin_shufflevector(b1lo, b1hi, 0,1,2,3,4,5,6,7,8,9,10,11,12,13,14,15);

        #pragma unroll
        for (int i = 0; i < 2; ++i)
            #pragma unroll
            for (int j = 0; j < 2; ++j)
                cacc[i][j] = __builtin_amdgcn_wmma_f32_16x16x32_f16(
                    false, af[i], false, bf[j], (short)0, cacc[i][j], false, false);

        __syncthreads();   // next buffer fully staged, current reads complete
    }

    // epilogue (C/D layout: lane 0-15 -> M=r, lane 16-31 -> M=8+r; N = lane&15)
    #pragma unroll
    for (int i = 0; i < 2; ++i)
        #pragma unroll
        for (int j = 0; j < 2; ++j)
            #pragma unroll
            for (int r = 0; r < 8; ++r) {
                int grow = tileM * 128 + waveM * 32 + i * 16 + lhalf * 8 + r;
                int gcol = tileN * 64  + waveN * 32 + j * 16 + lmod;
                if (grow < M) {
                    float v = cacc[i][j][r];
                    if (EPI == EPI_BIAS || EPI == EPI_GELU || EPI == EPI_RES) v += bias[gcol];
                    if (EPI == EPI_GELU) v = 0.5f * v * (1.0f + erff(v * 0.70710678118654752f));
                    if (EPI == EPI_RES) v += Rb[(size_t)grow * N + gcol];
                    if (EPI == EPI_SCORES) {
                        v += Rb[(size_t)grow * N + gcol];
                        if (grow == gcol) v = NEGC;
                    }
                    Cb[(size_t)grow * N + gcol] = v;
                }
            }
}

// ======================= top-k routing (one wave / row) =======================
__global__ __launch_bounds__(32)
void topk_kernel(const float* __restrict__ scores, int* __restrict__ routes,
                 float* __restrict__ rbias)
{
    const int row  = blockIdx.x;          // b*P + p
    const int lane = threadIdx.x;
    const float* sr = scores + (size_t)row * Pp;

    unsigned picked = 0;                  // 18 strided elems per lane
    float selv = 0.f; int seli = 0;
    for (int it = 0; it < KWw; ++it) {
        float lm = -3e38f; int lgi = 0;
        #pragma unroll
        for (int t = 0; t < 18; ++t) {
            int gi = lane + t * 32;
            float v = ((picked >> t) & 1u) ? -3e38f : sr[gi] * TEMP_INV;
            if (v > lm) { lm = v; lgi = gi; }
        }
        for (int off = 16; off > 0; off >>= 1) {   // wave argmax butterfly
            float ov = __shfl_xor(lm, off, 32);
            int   oi = __shfl_xor(lgi, off, 32);
            if (ov > lm || (ov == lm && oi < lgi)) { lm = ov; lgi = oi; }
        }
        if ((lgi & 31) == lane) picked |= 1u << (lgi >> 5);
        if (lane == it) { selv = lm; seli = lgi; }
    }
    // softmax over the 32 selected values -> clamped log weights
    float m = selv;
    for (int off = 16; off > 0; off >>= 1) m = fmaxf(m, __shfl_xor(m, off, 32));
    float e = expf(selv - m);
    float s = e;
    for (int off = 16; off > 0; off >>= 1) s += __shfl_xor(s, off, 32);
    float lb = fmaxf((selv - m) - logf(s), -10.0f);
    routes[(size_t)row * KWw + lane] = seli;
    rbias[(size_t)row * KWw + lane] = lb;
}

// ======================= CLS-token full attention =======================
__global__ __launch_bounds__(64)
void cls_attn_kernel(const float* __restrict__ qkv, float* __restrict__ attnout)
{
    __shared__ float qs[64];
    __shared__ float sc[Ss];
    const int bh = blockIdx.x;
    const int b = bh / Hh, h = bh % Hh;
    const int t = threadIdx.x;
    const size_t rowbase = (size_t)b * Ss;

    qs[t] = qkv[rowbase * D3 + (size_t)h * 64 + t];
    __syncthreads();
    for (int kk = t; kk < Ss; kk += 64) {
        const float* kp = qkv + (rowbase + kk) * D3 + Dm + h * 64;
        float d = 0.f;
        #pragma unroll
        for (int c = 0; c < 64; ++c) d += qs[c] * kp[c];
        sc[kk] = d * SCALEa;
    }
    __syncthreads();
    float m = -3e38f;
    for (int kk = 0; kk < Ss; ++kk) m = fmaxf(m, sc[kk]);
    float s = 0.f;
    for (int kk = 0; kk < Ss; ++kk) s += expf(sc[kk] - m);
    const float inv = 1.0f / s;
    float acc = 0.f;
    for (int kk = 0; kk < Ss; ++kk)
        acc += expf(sc[kk] - m) * inv * qkv[(rowbase + kk) * D3 + 2 * Dm + h * 64 + t];
    attnout[rowbase * Dm + (size_t)h * 64 + t] = acc;
}

// ======================= routed 32-key patch attention (wave per (b,h,p)) =======================
__global__ __launch_bounds__(256)
void patch_attn_kernel(const float* __restrict__ qkv, const int* __restrict__ routes,
                       const float* __restrict__ rbias, float* __restrict__ attnout)
{
    const int gw   = blockIdx.x * 8 + (threadIdx.x >> 5);
    const int lane = threadIdx.x & 31;
    if (gw >= Bb * Hh * Pp) return;
    const int p = gw % Pp;
    const int h = (gw / Pp) % Hh;
    const int b = gw / (Pp * Hh);

    const size_t qrow = (size_t)b * Ss + 1 + p;
    const float* qb = qkv + qrow * D3 + h * 64;
    const int rid = routes[((size_t)b * Pp + p) * KWw + lane];
    const float* kb = qkv + ((size_t)b * Ss + 1 + rid) * D3 + Dm + h * 64;

    float s = 0.f;
    #pragma unroll
    for (int c = 0; c < 64; ++c) s += qb[c] * kb[c];
    s = s * SCALEa + rbias[((size_t)b * Pp + p) * KWw + lane];

    float m = s;
    for (int off = 16; off > 0; off >>= 1) m = fmaxf(m, __shfl_xor(m, off, 32));
    float e = expf(s - m);
    float sum = e;
    for (int off = 16; off > 0; off >>= 1) sum += __shfl_xor(sum, off, 32);
    const float a = e / sum;

    const int d0 = lane * 2;
    float acc0 = 0.f, acc1 = 0.f;
    for (int j = 0; j < 32; ++j) {
        float aj = __shfl(a, j, 32);
        int   rj = __shfl(rid, j, 32);
        const float* vb = qkv + ((size_t)b * Ss + 1 + rj) * D3 + 2 * Dm + h * 64;
        acc0 += aj * vb[d0];
        acc1 += aj * vb[d0 + 1];
    }
    float* ob = attnout + qrow * Dm + h * 64;
    ob[d0]     = acc0;
    ob[d0 + 1] = acc1;
}

// ======================= launcher =======================
extern "C" void kernel_launch(void* const* d_in, const int* in_sizes, int n_in,
                              void* d_out, int out_size, void* d_ws, size_t ws_size,
                              hipStream_t stream)
{
    (void)in_sizes; (void)n_in; (void)out_size; (void)ws_size;
    const float* x    = (const float*)d_in[0];
    const float* Wq   = (const float*)d_in[1];
    const float* bq   = (const float*)d_in[2];
    const float* Wk   = (const float*)d_in[3];
    const float* bk   = (const float*)d_in[4];
    const float* posb = (const float*)d_in[5];
    const float* g1   = (const float*)d_in[6];
    const float* b1n  = (const float*)d_in[7];
    const float* Wqkv = (const float*)d_in[8];
    const float* bqkv = (const float*)d_in[9];
    const float* Wp   = (const float*)d_in[10];
    const float* bp   = (const float*)d_in[11];
    const float* g2   = (const float*)d_in[12];
    const float* b2n  = (const float*)d_in[13];
    const float* W1   = (const float*)d_in[14];
    const float* bm1  = (const float*)d_in[15];
    const float* W2   = (const float*)d_in[16];
    const float* bm2  = (const float*)d_in[17];
    float* out = (float*)d_out;
    float* ws  = (float*)d_ws;

    const size_t BS = (size_t)BSr; // 4616
    float* xn  = ws;
    float* qf  = xn  + BS * Dm;
    float* kf  = qf  + BS * Dm;
    float* scr = kf  + BS * Dm;
    float* qkv = scr + (size_t)Bb * Pp * Pp;
    float* att = qkv + BS * D3;
    float* x1  = att + BS * Dm;
    float* h2  = x1  + BS * Dm;
    int*   routes = (int*)(h2 + BS * Dm);
    float* rbias  = (float*)(routes + (size_t)Bb * Pp * KWw);
    float* m1 = qf;   // MLP hidden aliases dead q/k/scores/qkv region

    const dim3 blk(256);
    const long long bstride = (long long)Ss * Dm;   // per-batch row stride in q/k
    const long long sstride = (long long)Pp * Pp;   // per-batch score stride

    // 1) pre-LN
    ln_kernel<<<dim3(BSr), blk, 0, stream>>>(x, g1, b1n, xn);

    // 2) routing projections (all rows; cls rows unused downstream)
    gemm_wmma_kernel<EPI_BIAS, false><<<dim3(12, 37, 1), blk, 0, stream>>>(
        xn, Wq, bq, nullptr, qf, BSr, Dm, Dm, 0, 0, 0, 0);
    gemm_wmma_kernel<EPI_BIAS, false><<<dim3(12, 37, 1), blk, 0, stream>>>(
        xn, Wk, bk, nullptr, kf, BSr, Dm, Dm, 0, 0, 0, 0);
    l2norm_kernel<<<dim3(BSr), blk, 0, stream>>>(qf);
    l2norm_kernel<<<dim3(BSr), blk, 0, stream>>>(kf);

    // 3) routing scores: q_p @ k_p^T + pos_bias, diag = NEG  (batched, B transposed)
    gemm_wmma_kernel<EPI_SCORES, true><<<dim3(9, 5, Bb), blk, 0, stream>>>(
        qf + Dm, kf + Dm, nullptr, posb, scr, Pp, Pp, Dm, bstride, bstride, sstride, 0);

    // 4) top-32 + routing softmax
    topk_kernel<<<dim3(Bb * Pp), dim3(32), 0, stream>>>(scr, routes, rbias);

    // 5) fused QKV projection
    gemm_wmma_kernel<EPI_BIAS, false><<<dim3(36, 37, 1), blk, 0, stream>>>(
        xn, Wqkv, bqkv, nullptr, qkv, BSr, D3, Dm, 0, 0, 0, 0);

    // 6) attention
    cls_attn_kernel<<<dim3(Bb * Hh), dim3(64), 0, stream>>>(qkv, att);
    patch_attn_kernel<<<dim3((Bb * Hh * Pp) / 8), blk, 0, stream>>>(qkv, routes, rbias, att);

    // 7) output projection + residual
    gemm_wmma_kernel<EPI_RES, false><<<dim3(12, 37, 1), blk, 0, stream>>>(
        att, Wp, bp, x, x1, BSr, Dm, Dm, 0, 0, 0, 0);

    // 8) MLP
    ln_kernel<<<dim3(BSr), blk, 0, stream>>>(x1, g2, b2n, h2);
    gemm_wmma_kernel<EPI_GELU, false><<<dim3(48, 37, 1), blk, 0, stream>>>(
        h2, W1, bm1, nullptr, m1, BSr, MLPm, Dm, 0, 0, 0, 0);
    gemm_wmma_kernel<EPI_RES, false><<<dim3(12, 37, 1), blk, 0, stream>>>(
        m1, W2, bm2, x1, out, BSr, Dm, MLPm, 0, 0, 0, 0);
}
